// MultiheadAttention_29506425324122
// MI455X (gfx1250) — matrix operand order repaired and checked by reading, not compiled, not run on hardware
//
#include <hip/hip_runtime.h>

// Problem constants (match reference)
#define SDIM 2048
#define EDIM 512
#define NH   8
#define DHD  64
#define NBH  16   // B * H
#define NB   2

typedef _Float16 v16h __attribute__((ext_vector_type(16)));
typedef float    v8f  __attribute__((ext_vector_type(8)));

__device__ __forceinline__ v8f wmma_f16(v16h a, v16h b, v8f c) {
  // D = A(16x32 f16) x B(32x16 f16) + C(16x16 f32)
  return __builtin_amdgcn_wmma_f32_16x16x32_f16(
      /*neg_a=*/false, a, /*neg_b=*/false, b,
      /*c_mod=*/(short)0, c, /*reuse_a=*/false, /*reuse_b=*/false);
}

// ---------------------------------------------------------------------------
// Kernel 1: input projection  Y = X @ W^T + b   (fp32 in -> f16 out,
// output written directly in head-batched layout [B*H, S, DH])
// grid = (EDIM/64, (NB*SDIM)/64), block = 128 (4 waves), each wave: 16x64 tile
// ---------------------------------------------------------------------------
__global__ __launch_bounds__(128)
void proj_kernel(const float* __restrict__ X, const float* __restrict__ W,
                 const float* __restrict__ bias, _Float16* __restrict__ Y) {
  const int tid  = threadIdx.x;
  const int wave = tid >> 5;
  const int lane = tid & 31;
  const int hi   = lane >> 4;      // lane half
  const int lid  = lane & 15;
  const int m0 = blockIdx.y * 64 + wave * 16;   // rows of X (B*S)
  const int n0 = blockIdx.x * 64;               // cols of Y (E)

  v8f acc[4] = {};
  const float* arow = X + (size_t)(m0 + lid) * EDIM;

  for (int kc = 0; kc < EDIM; kc += 32) {
    v16h a;
    const float* ap = arow + kc;
#pragma unroll
    for (int j = 0; j < 8; ++j) a[j]     = (_Float16)ap[hi * 16 + j];
#pragma unroll
    for (int j = 0; j < 8; ++j) a[8 + j] = (_Float16)ap[hi * 16 + 8 + j];
#pragma unroll
    for (int t = 0; t < 4; ++t) {
      // B[k][n] = W[n][k]  (x @ W^T): lane owns column n0+t*16+lid
      const float* bp = W + (size_t)(n0 + t * 16 + lid) * EDIM + kc + hi * 16;
      v16h b;
#pragma unroll
      for (int j = 0; j < 16; ++j) b[j] = (_Float16)bp[j];
      acc[t] = wmma_f16(a, b, acc[t]);
    }
  }
#pragma unroll
  for (int t = 0; t < 4; ++t) {
    const int n = n0 + t * 16 + lid;
    const int h = n >> 6, d = n & 63;
    const float bv = bias[n];
#pragma unroll
    for (int v = 0; v < 8; ++v) {
      const int m = m0 + hi * 8 + v;
      const int bb = m >> 11, s = m & 2047;
      Y[((size_t)(bb * NH + h) * SDIM + s) * DHD + d] = (_Float16)(acc[t][v] + bv);
    }
  }
}

// ---------------------------------------------------------------------------
// Kernel 2 (fused): scores + softmax + attn @ V.
// Block = 256 threads (8 waves) handles 16 query rows of one head.
//  Phase 1: wave w computes scores[16 x 256] for cols [w*256, w*256+256) in
//           registers (16 v8f accumulators, WMMA f16).
//  Phase 2: softmax across registers: shfl_xor within lane-halves + LDS
//           cross-wave reduce. Normalized fp32 probs -> d_out via NT stores.
//  Phase 3: each wave multiplies its own prob slice (C-layout -> A-layout via
//           small per-wave LDS bounce) against V rows [w*256,+256); partials
//           reduced with ds_add_f32 atomics, written as f16 [B,S,E].
// grid = (S/16, NBH)
// ---------------------------------------------------------------------------
__global__ __launch_bounds__(256)
void fused_attn_kernel(const _Float16* __restrict__ Qh,
                       const _Float16* __restrict__ Kh,
                       const _Float16* __restrict__ Vh,
                       float* __restrict__ attn,
                       _Float16* __restrict__ Oh) {
  __shared__ float    smax[8][16];
  __shared__ float    ssum[8][16];
  __shared__ _Float16 abuf[8][16][40];   // per-wave 16x32 transpose staging
  __shared__ float    osum[16][DHD];     // cross-wave output accumulator

  const int tid  = threadIdx.x;
  const int wave = tid >> 5;
  const int lane = tid & 31;
  const int hi   = lane >> 4;
  const int lid  = lane & 15;
  const int bh = blockIdx.y;
  const int m0 = blockIdx.x * 16;        // 16 query rows per block
  const int c0 = wave * 256;             // this wave's key-column base

  // zero the shared output accumulator (first barrier below orders this)
  for (int i = tid; i < 16 * DHD; i += 256) (&osum[0][0])[i] = 0.0f;

  const _Float16* qb = Qh + (size_t)bh * SDIM * DHD;
  const _Float16* kb = Kh + (size_t)bh * SDIM * DHD;

  // ---- Phase 1: scores (16 rows x 256 cols per wave) --------------------
  v16h a0, a1;
  {
    const _Float16* ap = qb + (size_t)(m0 + lid) * DHD;
#pragma unroll
    for (int j = 0; j < 8; ++j) { a0[j] = ap[hi*16+j];    a0[8+j] = ap[hi*16+8+j]; }
#pragma unroll
    for (int j = 0; j < 8; ++j) { a1[j] = ap[32+hi*16+j]; a1[8+j] = ap[40+hi*16+j]; }
  }
  v8f acc[16] = {};
#pragma unroll
  for (int t = 0; t < 16; ++t) {
    const _Float16* bp = kb + (size_t)(c0 + t * 16 + lid) * DHD;
    v16h b0, b1;
#pragma unroll
    for (int j = 0; j < 16; ++j) { b0[j] = bp[hi*16+j]; b1[j] = bp[32+hi*16+j]; }
    acc[t] = wmma_f16(a0, b0, acc[t]);
    acc[t] = wmma_f16(a1, b1, acc[t]);
  }
#pragma unroll
  for (int t = 0; t < 16; ++t)
#pragma unroll
    for (int v = 0; v < 8; ++v) acc[t][v] *= 0.125f;   // 1/sqrt(64)

  // ---- Phase 2: softmax --------------------------------------------------
  float rmax[8];
#pragma unroll
  for (int v = 0; v < 8; ++v) {
    float m = acc[0][v];
#pragma unroll
    for (int t = 1; t < 16; ++t) m = fmaxf(m, acc[t][v]);
#pragma unroll
    for (int off = 8; off > 0; off >>= 1) m = fmaxf(m, __shfl_xor(m, off, 32));
    rmax[v] = m;
  }
  if (lid == 0) {
#pragma unroll
    for (int v = 0; v < 8; ++v) smax[wave][hi * 8 + v] = rmax[v];
  }
  __syncthreads();
  float M[8];
#pragma unroll
  for (int v = 0; v < 8; ++v) {
    float m = smax[0][hi * 8 + v];
#pragma unroll
    for (int w = 1; w < 8; ++w) m = fmaxf(m, smax[w][hi * 8 + v]);
    M[v] = m;
  }
  float rsum[8];
#pragma unroll
  for (int v = 0; v < 8; ++v) rsum[v] = 0.0f;
#pragma unroll
  for (int t = 0; t < 16; ++t)
#pragma unroll
    for (int v = 0; v < 8; ++v) {
      const float p = __expf(acc[t][v] - M[v]);
      acc[t][v] = p;
      rsum[v] += p;
    }
#pragma unroll
  for (int v = 0; v < 8; ++v) {
#pragma unroll
    for (int off = 8; off > 0; off >>= 1) rsum[v] += __shfl_xor(rsum[v], off, 32);
  }
  if (lid == 0) {
#pragma unroll
    for (int v = 0; v < 8; ++v) ssum[wave][hi * 8 + v] = rsum[v];
  }
  __syncthreads();
  float inv[8];
#pragma unroll
  for (int v = 0; v < 8; ++v) {
    float s = ssum[0][hi * 8 + v];
#pragma unroll
    for (int w = 1; w < 8; ++w) s += ssum[w][hi * 8 + v];
    inv[v] = 1.0f / s;
  }

  // normalized fp32 probabilities -> d_out (streaming, non-temporal)
  float* ob = attn + (size_t)bh * SDIM * SDIM;
#pragma unroll
  for (int t = 0; t < 16; ++t) {
#pragma unroll
    for (int v = 0; v < 8; ++v) {
      __builtin_nontemporal_store(
          acc[t][v] * inv[v],
          &ob[(size_t)(m0 + hi * 8 + v) * SDIM + c0 + t * 16 + lid]);
    }
  }

  // ---- Phase 3: this wave's prob slice @ V -------------------------------
  const _Float16* vb = Vh + (size_t)bh * SDIM * DHD + (size_t)c0 * DHD;
  _Float16 (*aw)[40] = abuf[wave];
  v8f oacc[4] = {};
#pragma unroll
  for (int kc = 0; kc < 256; kc += 32) {
    const int tbase = kc >> 4;
    // C-layout (lane = col) -> A-layout (lane = row) via per-wave LDS bounce;
    // same-wave DS ops are in-order, no barrier required.
#pragma unroll
    for (int tt = 0; tt < 2; ++tt)
#pragma unroll
      for (int v = 0; v < 8; ++v)
        aw[hi * 8 + v][tt * 16 + lid] = (_Float16)(acc[tbase + tt][v] * inv[v]);
    v16h a;
#pragma unroll
    for (int j = 0; j < 8; ++j) { a[j] = aw[lid][hi*16+j]; a[8+j] = aw[lid][hi*16+8+j]; }
#pragma unroll
    for (int t = 0; t < 4; ++t) {
      v16h b;
#pragma unroll
      for (int j = 0; j < 16; ++j)
        b[j] = vb[(size_t)(kc + hi * 16 + j) * DHD + t * 16 + lid];
      oacc[t] = wmma_f16(a, b, oacc[t]);
    }
  }
  // reduce wave partials into shared accumulator (ds_add_f32)
#pragma unroll
  for (int t = 0; t < 4; ++t)
#pragma unroll
    for (int v = 0; v < 8; ++v)
      atomicAdd(&osum[hi * 8 + v][t * 16 + lid], oacc[t][v]);
  __syncthreads();

  const int bb = bh / NH, h = bh % NH;
  for (int i = tid; i < 16 * DHD; i += 256) {
    const int r = i >> 6, c = i & 63;
    Oh[((size_t)(bb * SDIM + m0 + r)) * EDIM + h * DHD + c] = (_Float16)osum[r][c];
  }
}

// ---------------------------------------------------------------------------
// Kernel 3: out = Oh @ Wo^T + bo   (f16 A, fp32 W -> fp32 out in d_out)
// grid = (EDIM/64, (NB*SDIM)/64), block = 128
// ---------------------------------------------------------------------------
__global__ __launch_bounds__(128)
void oproj_kernel(const _Float16* __restrict__ Oh, const float* __restrict__ Wo,
                  const float* __restrict__ bo, float* __restrict__ out) {
  const int tid  = threadIdx.x;
  const int wave = tid >> 5;
  const int lane = tid & 31;
  const int hi   = lane >> 4;
  const int lid  = lane & 15;
  const int m0 = blockIdx.y * 64 + wave * 16;
  const int n0 = blockIdx.x * 64;

  v8f acc[4] = {};
  const _Float16* arow = Oh + (size_t)(m0 + lid) * EDIM;

  for (int kc = 0; kc < EDIM; kc += 32) {
    v16h a;
    const _Float16* ap = arow + kc;
#pragma unroll
    for (int j = 0; j < 8; ++j) a[j]     = ap[hi * 16 + j];
#pragma unroll
    for (int j = 0; j < 8; ++j) a[8 + j] = ap[hi * 16 + 8 + j];
#pragma unroll
    for (int t = 0; t < 4; ++t) {
      const float* bp = Wo + (size_t)(n0 + t * 16 + lid) * EDIM + kc + hi * 16;
      v16h b;
#pragma unroll
      for (int j = 0; j < 16; ++j) b[j] = (_Float16)bp[j];
      acc[t] = wmma_f16(a, b, acc[t]);
    }
  }
#pragma unroll
  for (int t = 0; t < 4; ++t) {
    const int n = n0 + t * 16 + lid;
    const float bv = bo[n];
#pragma unroll
    for (int v = 0; v < 8; ++v) {
      const int m = m0 + hi * 8 + v;
      out[(size_t)m * EDIM + n] = acc[t][v] + bv;
    }
  }
}

// ---------------------------------------------------------------------------
extern "C" void kernel_launch(void* const* d_in, const int* in_sizes, int n_in,
                              void* d_out, int out_size, void* d_ws, size_t ws_size,
                              hipStream_t stream) {
  (void)in_sizes; (void)n_in; (void)out_size; (void)ws_size;
  const float* q  = (const float*)d_in[0];
  const float* k  = (const float*)d_in[1];
  const float* v  = (const float*)d_in[2];
  const float* Wq = (const float*)d_in[3];
  const float* bq = (const float*)d_in[4];
  const float* Wk = (const float*)d_in[5];
  const float* bk = (const float*)d_in[6];
  const float* Wv = (const float*)d_in[7];
  const float* bv = (const float*)d_in[8];
  const float* Wo = (const float*)d_in[9];
  const float* bo = (const float*)d_in[10];

  float* out  = (float*)d_out;                       // [B,S,E]
  float* attn = out + (size_t)NB * SDIM * EDIM;      // [B*H,S,S]

  _Float16* Qh = (_Float16*)d_ws;                    // [B*H,S,DH] f16
  _Float16* Kh = Qh + (size_t)NBH * SDIM * DHD;
  _Float16* Vh = Kh + (size_t)NBH * SDIM * DHD;
  _Float16* Oh = Vh + (size_t)NBH * SDIM * DHD;      // [B,S,E] f16

  dim3 gproj(EDIM / 64, (NB * SDIM) / 64);           // (8, 64)
  proj_kernel<<<gproj, 128, 0, stream>>>(q, Wq, bq, Qh);
  proj_kernel<<<gproj, 128, 0, stream>>>(k, Wk, bk, Kh);
  proj_kernel<<<gproj, 128, 0, stream>>>(v, Wv, bv, Vh);

  fused_attn_kernel<<<dim3(SDIM / 16, NBH), 256, 0, stream>>>(Qh, Kh, Vh, attn, Oh);

  oproj_kernel<<<gproj, 128, 0, stream>>>(Oh, Wo, bo, out);
}
